// FlaxPhiAttention_75608604279460
// MI455X (gfx1250) — compile-verified, matrix operand order checked by reading, not run on hardware
//
#include <hip/hip_runtime.h>
#include <hip/hip_bf16.h>
#include <cstddef>

// ---------------------------------------------------------------------------
// Types for CDNA5 WMMA (gfx1250, wave32): V_WMMA_F32_16X16X32_BF16
// ---------------------------------------------------------------------------
typedef __bf16 bf16;
typedef __attribute__((ext_vector_type(16))) __bf16 v16bf;
typedef __attribute__((ext_vector_type(8)))  __bf16 v8bf;
typedef __attribute__((ext_vector_type(2)))  __bf16 v2bf;
typedef __attribute__((ext_vector_type(8)))  float  v8f;

union Frag {
    v16bf        v;
    v8bf         h[2];
    unsigned int u[8];
};

// Native fptrunc (RNE) -> v_cvt_pk_bf16_f32 pairs on gfx1250
__device__ __forceinline__ bf16 to_bf16(float f) { return (bf16)f; }

__device__ __forceinline__ unsigned int pack2_bf16(float a, float b) {
    v2bf p;
    p.x = (bf16)a;
    p.y = (bf16)b;
    return __builtin_bit_cast(unsigned int, p);
}

__device__ __forceinline__ v8f wmma_bf16(const Frag& a, const Frag& b, v8f c) {
    return __builtin_amdgcn_wmma_f32_16x16x32_bf16(
        /*neg_a=*/false, a.v, /*neg_b=*/false, b.v,
        /*c_mod=*/(short)0, c, /*reuse_a=*/false, /*reuse_b=*/false);
}

// ---------------------------------------------------------------------------
// Problem dimensions (compile-time so all address math folds to immediates)
// ---------------------------------------------------------------------------
#define HID 2560
#define NH 32
#define HD 80
#define RD 32
#define SEQ 2048
#define ROWS 4096              // B * SEQ

// ---------------------------------------------------------------------------
// Kernel 1: C[M,N] = A[M,K] @ W[K,N] + bias, fp32 in/out, bf16 WMMA compute.
// M = ROWS, N = K = HID (compile-time).  Block tile 128x128, BK=32,
// 256 threads = 8 waves (4 M-waves x 2 N-waves), 2x4 accumulators per wave.
// ---------------------------------------------------------------------------
#define BM 128
#define BN 128
#define BK 32

__global__ __launch_bounds__(256) void gemm_bf16(const float* __restrict__ X,
                                                 const float* __restrict__ W,
                                                 const float* __restrict__ bias,
                                                 float* __restrict__ out) {
    constexpr int N = HID;
    constexpr int K = HID;

    __shared__ alignas(16) bf16 sA[BM][BK];   // X tile, row-major [m][k]
    __shared__ alignas(16) bf16 sB[BN][BK];   // W tile transposed: sB[n][k] = W[k][n]

    const int tid  = threadIdx.x;
    const int lane = tid & 31;
    const int wave = tid >> 5;
    const int wm   = wave & 3;        // 0..3 -> 32-row slice
    const int wn   = wave >> 2;       // 0..1 -> 64-col slice
    const int hi   = (lane >> 4) & 1;
    const int l16  = lane & 15;

    const int m0 = blockIdx.y * BM;
    const int n0 = blockIdx.x * BN;

    // A-stage mapping: thread t -> row t/2, 16-float segment t%2
    const int arow = tid >> 1;
    const int aseg = (tid & 1) * 16;
    // B-stage mapping: thread t -> column n = t%128, 16-k segment (t/128)*16
    const int bn  = tid & 127;
    const int bks = (tid >> 7) * 16;

    const float* aptr = &X[(size_t)(m0 + arow) * K + aseg];
    const float* bptr = &W[(size_t)bks * N + n0 + bn];

    v8f acc[2][4];
#pragma unroll
    for (int i = 0; i < 2; ++i)
#pragma unroll
        for (int j = 0; j < 4; ++j) acc[i][j] = {};

    for (int k0 = 0; k0 < K; k0 += BK) {
        // ---- stage A (128x32 fp32 -> bf16 LDS), 2x ds_store_b128 ----
        {
            const float4* ap = (const float4*)(aptr + k0);
            float4 f0 = ap[0], f1 = ap[1], f2 = ap[2], f3 = ap[3];
            uint4 w0, w1;
            w0.x = pack2_bf16(f0.x, f0.y); w0.y = pack2_bf16(f0.z, f0.w);
            w0.z = pack2_bf16(f1.x, f1.y); w0.w = pack2_bf16(f1.z, f1.w);
            w1.x = pack2_bf16(f2.x, f2.y); w1.y = pack2_bf16(f2.z, f2.w);
            w1.z = pack2_bf16(f3.x, f3.y); w1.w = pack2_bf16(f3.z, f3.w);
            *(uint4*)&sA[arow][aseg]     = w0;
            *(uint4*)&sA[arow][aseg + 8] = w1;
        }
        // ---- stage B transposed: thread owns column n, reads 16 k-values.
        //      N is compile-time -> one base reg + immediate offsets (clause),
        //      coalesced across lanes; stores 2x ds_store_b128 ----
        {
            const float* wp = bptr + (size_t)k0 * N;
            float v[16];
#pragma unroll
            for (int i = 0; i < 16; ++i) v[i] = wp[i * N];
            uint4 w0, w1;
            w0.x = pack2_bf16(v[0],  v[1]);  w0.y = pack2_bf16(v[2],  v[3]);
            w0.z = pack2_bf16(v[4],  v[5]);  w0.w = pack2_bf16(v[6],  v[7]);
            w1.x = pack2_bf16(v[8],  v[9]);  w1.y = pack2_bf16(v[10], v[11]);
            w1.z = pack2_bf16(v[12], v[13]); w1.w = pack2_bf16(v[14], v[15]);
            *(uint4*)&sB[bn][bks]     = w0;
            *(uint4*)&sB[bn][bks + 8] = w1;
        }
        if (k0 + BK < K) {
            __builtin_prefetch(aptr + k0 + BK, 0, 1);
            __builtin_prefetch(bptr + (size_t)(k0 + BK) * N, 0, 1);
        }
        __syncthreads();

        // ---- fragments (two aligned 16B chunks per lane, K-pattern layout) ----
        Frag fa[2], fb[4];
#pragma unroll
        for (int mt = 0; mt < 2; ++mt) {
            const int m = wm * 32 + mt * 16 + l16;
            fa[mt].h[0] = *(const v8bf*)&sA[m][hi * 8];
            fa[mt].h[1] = *(const v8bf*)&sA[m][16 + hi * 8];
        }
#pragma unroll
        for (int nt = 0; nt < 4; ++nt) {
            const int n = wn * 64 + nt * 16 + l16;
            fb[nt].h[0] = *(const v8bf*)&sB[n][hi * 8];
            fb[nt].h[1] = *(const v8bf*)&sB[n][16 + hi * 8];
        }
#pragma unroll
        for (int mt = 0; mt < 2; ++mt)
#pragma unroll
            for (int nt = 0; nt < 4; ++nt)
                acc[mt][nt] = wmma_bf16(fa[mt], fb[nt], acc[mt][nt]);
        __syncthreads();
    }

    // ---- epilogue: C layout row = r + 8*hi, col = l16 ----
#pragma unroll
    for (int mt = 0; mt < 2; ++mt) {
#pragma unroll
        for (int nt = 0; nt < 4; ++nt) {
            const int col = n0 + wn * 64 + nt * 16 + l16;
            const float b = bias[col];
#pragma unroll
            for (int r = 0; r < 8; ++r) {
                const int row = m0 + wm * 32 + mt * 16 + r + 8 * hi;
                out[(size_t)row * N + col] = acc[mt][nt][r] + b;
            }
        }
    }
}

// ---------------------------------------------------------------------------
// Kernel 2: per-row LayerNorm (over 2560) + partial RoPE (first 32 of each 80)
// + pack fp32 -> bf16 in [B, H, S, D] layout.  One block (256 thr) per row.
// ---------------------------------------------------------------------------
__global__ __launch_bounds__(256) void ln_rope_pack(const float* __restrict__ src,
                                                    const float* __restrict__ scale,
                                                    const float* __restrict__ bias,
                                                    const int* __restrict__ pos_ids,
                                                    const float* __restrict__ sin_t,
                                                    const float* __restrict__ cos_t,
                                                    bf16* __restrict__ dst) {
    __shared__ float srow[HID];
    __shared__ float w1[8], w2[8];

    const int row  = blockIdx.x;          // b*SEQ + s
    const int tid  = threadIdx.x;
    const int lane = tid & 31;
    const int wave = tid >> 5;
    const int b = row >> 11;
    const int s = row & (SEQ - 1);

    float s1 = 0.f, s2 = 0.f;
    for (int c = tid; c < HID; c += 256) {
        float x = src[(size_t)row * HID + c];
        srow[c] = x;
        s1 += x;
        s2 += x * x;
    }
#pragma unroll
    for (int m = 16; m >= 1; m >>= 1) {
        s1 += __shfl_xor(s1, m, 32);
        s2 += __shfl_xor(s2, m, 32);
    }
    if (lane == 0) { w1[wave] = s1; w2[wave] = s2; }
    __syncthreads();
    float t1 = 0.f, t2 = 0.f;
#pragma unroll
    for (int i = 0; i < 8; ++i) { t1 += w1[i]; t2 += w2[i]; }
    const float mu   = t1 * (1.0f / HID);
    const float var  = t2 * (1.0f / HID) - mu * mu;
    const float rstd = rsqrtf(var + 1e-5f);

    for (int c = tid; c < HID; c += 256)
        srow[c] = (srow[c] - mu) * rstd * scale[c] + bias[c];
    __syncthreads();

    const int p = pos_ids[row];
    for (int c = tid; c < HID; c += 256) {
        const int h = c / HD;
        const int d = c % HD;
        float val = srow[c];
        if (d < RD) {
            const float cs = cos_t[p * RD + d];
            const float sn = sin_t[p * RD + d];
            const float other = srow[h * HD + ((d < RD / 2) ? d + RD / 2 : d - RD / 2)];
            val = val * cs + ((d < RD / 2) ? -other : other) * sn;
        }
        dst[(((size_t)(b * NH + h)) * SEQ + s) * HD + d] = to_bf16(val);
    }
}

// Kernel 3: V fp32 [B*S, HID] -> bf16 [B, H, S, D]
__global__ __launch_bounds__(256) void v_pack(const float* __restrict__ src,
                                              bf16* __restrict__ dst) {
    const int idx = blockIdx.x * 256 + threadIdx.x;   // grid sized exactly
    const int row = idx / HID;            // b*SEQ + s
    const int c   = idx - row * HID;
    const int b = row >> 11, s = row & (SEQ - 1);
    const int h = c / HD, d = c % HD;
    dst[(((size_t)(b * NH + h)) * SEQ + s) * HD + d] = to_bf16(src[idx]);
}

// ---------------------------------------------------------------------------
// Kernel 4: flash-style causal attention. Grid (S/64, B*H), 128 thr = 4 waves.
// Each wave owns a 16-row Q tile; scores & P*V via bf16 WMMA, softmax in fp32.
// ---------------------------------------------------------------------------
__device__ __forceinline__ void load_frag_row80(Frag& f, const bf16* base,
                                                int row, int t, int hi) {
    // t = 0,1: full 32-d chunk at d0 = 32*t ; t = 2: half chunk d 64..79, rest 0
    if (t < 2) {
        const bf16* p = base + (size_t)row * HD + t * 32 + hi * 8;
        f.h[0] = *(const v8bf*)p;
        f.h[1] = *(const v8bf*)(p + 16);
    } else {
        const bf16* p = base + (size_t)row * HD + 64 + hi * 8;
        f.h[0] = *(const v8bf*)p;
        f.u[4] = f.u[5] = f.u[6] = f.u[7] = 0;
    }
}

__global__ __launch_bounds__(128) void attention(const bf16* __restrict__ Qb,
                                                 const bf16* __restrict__ Kb,
                                                 const bf16* __restrict__ Vb,
                                                 const int* __restrict__ amask,
                                                 float* __restrict__ attnf) {
    __shared__ alignas(16) bf16 sP[4][16][16];    // per-wave P tile (C->A relayout)
    __shared__ alignas(16) bf16 sVT[4][HD][16];   // per-wave V^T tile: [d][key]

    const int lane = threadIdx.x & 31;
    const int wave = threadIdx.x >> 5;
    const int hi   = (lane >> 4) & 1;
    const int l16  = lane & 15;

    const int bh = blockIdx.y;        // b*NH + h
    const int b  = bh / NH;
    const int h  = bh - b * NH;
    const int q0 = blockIdx.x * 64 + wave * 16;

    const bf16* Qh = Qb + (size_t)bh * SEQ * HD;
    const bf16* Kh = Kb + (size_t)bh * SEQ * HD;
    const bf16* Vh = Vb + (size_t)bh * SEQ * HD;

    // Q fragments held for the whole K loop (3 frags: 32 + 32 + 16-padded)
    Frag fq[3];
#pragma unroll
    for (int t = 0; t < 3; ++t) load_frag_row80(fq[t], Qh, q0 + l16, t, hi);

    v8f oacc[5];
#pragma unroll
    for (int dt = 0; dt < 5; ++dt) oacc[dt] = {};
    float mrow[8], lrow[8];
#pragma unroll
    for (int r = 0; r < 8; ++r) { mrow[r] = -1e30f; lrow[r] = 0.f; }

    const float SCALE = 0.11180339887498948f;   // 1/sqrt(80)
    const int nkb = blockIdx.x * 4 + 4;         // causal: keys up to q0_block+63

    for (int kb = 0; kb < nkb; ++kb) {
        const int k0 = kb * 16;

        // ---- stage V^T tile for this wave: sVT[d][key] ----
        for (int idx = lane; idx < 16 * HD; idx += 32) {
            const int key = idx / HD;
            const int d   = idx - key * HD;
            sVT[wave][d][key] = Vh[(size_t)(k0 + key) * HD + d];
        }

        // ---- scores: S = Q (16x80) x K^T, 3 WMMAs, B-frags from global ----
        v8f sacc = {};
#pragma unroll
        for (int t = 0; t < 3; ++t) {
            Frag fk;
            load_frag_row80(fk, Kh, k0 + l16, t, hi);
            sacc = wmma_bf16(fq[t], fk, sacc);
        }

        // ---- online softmax (per row = r + 8*hi, cols = 16-lane half) ----
        const int key  = k0 + l16;
        const bool pv  = (amask[b * SEQ + key] != 0);
#pragma unroll
        for (int r = 0; r < 8; ++r) {
            const int q = q0 + r + 8 * hi;
            const bool valid = pv && (key <= q);
            float sc = valid ? sacc[r] * SCALE : -1e30f;
            float mx = sc;
#pragma unroll
            for (int m = 8; m >= 1; m >>= 1) mx = fmaxf(mx, __shfl_xor(mx, m, 32));
            const float mnew  = fmaxf(mrow[r], mx);
            const float alpha = __expf(mrow[r] - mnew);
            float pr = valid ? __expf(sc - mnew) : 0.f;
            float ps = pr;
#pragma unroll
            for (int m = 8; m >= 1; m >>= 1) ps += __shfl_xor(ps, m, 32);
            lrow[r] = lrow[r] * alpha + ps;
            mrow[r] = mnew;
#pragma unroll
            for (int dt = 0; dt < 5; ++dt) oacc[dt][r] *= alpha;
            sP[wave][r + 8 * hi][l16] = to_bf16(pr);
        }

        // ---- O += P (16x16, zero-padded K) x V (16x80): 5 WMMAs ----
        Frag fp;
        fp.h[0] = *(const v8bf*)&sP[wave][l16][hi * 8];
        fp.u[4] = fp.u[5] = fp.u[6] = fp.u[7] = 0;
#pragma unroll
        for (int dt = 0; dt < 5; ++dt) {
            Frag fv;
            fv.h[0] = *(const v8bf*)&sVT[wave][dt * 16 + l16][hi * 8];
            fv.u[4] = fv.u[5] = fv.u[6] = fv.u[7] = 0;
            oacc[dt] = wmma_bf16(fp, fv, oacc[dt]);
        }
    }

    // ---- write [B,S,HID] fp32 (col = h*80 + d) ----
#pragma unroll
    for (int dt = 0; dt < 5; ++dt) {
        const int col = h * HD + dt * 16 + l16;
#pragma unroll
        for (int r = 0; r < 8; ++r) {
            const int q = q0 + r + 8 * hi;
            const float inv = (lrow[r] > 0.f) ? (1.0f / lrow[r]) : 0.f;
            attnf[((size_t)(b * SEQ + q)) * HID + col] = oacc[dt][r] * inv;
        }
    }
}

// ---------------------------------------------------------------------------
// Launcher
// ---------------------------------------------------------------------------
extern "C" void kernel_launch(void* const* d_in, const int* in_sizes, int n_in,
                              void* d_out, int out_size, void* d_ws, size_t ws_size,
                              hipStream_t stream) {
    const float* hs    = (const float*)d_in[0];
    const int*   am    = (const int*)d_in[1];
    const int*   pos   = (const int*)d_in[2];
    const float* sin_t = (const float*)d_in[3];
    const float* cos_t = (const float*)d_in[4];
    const float* Wq = (const float*)d_in[5];
    const float* bq = (const float*)d_in[6];
    const float* Wk = (const float*)d_in[7];
    const float* bk = (const float*)d_in[8];
    const float* Wv = (const float*)d_in[9];
    const float* bv = (const float*)d_in[10];
    const float* Wo = (const float*)d_in[11];
    const float* bo = (const float*)d_in[12];
    const float* qls = (const float*)d_in[13];
    const float* qlb = (const float*)d_in[14];
    const float* kls = (const float*)d_in[15];
    const float* klb = (const float*)d_in[16];
    float* out = (float*)d_out;

    char* ws = (char*)d_ws;
    const size_t F32B = (size_t)ROWS * HID * sizeof(float); // 41,943,040
    const size_t B16B = (size_t)ROWS * HID * sizeof(bf16);  // 20,971,520
    float* qf = (float*)(ws);
    float* kf = (float*)(ws + F32B);
    float* vf = (float*)(ws + 2 * F32B);
    bf16*  Qb = (bf16*)(ws + 3 * F32B);
    bf16*  Kb = (bf16*)(ws + 3 * F32B + B16B);
    bf16*  Vb = (bf16*)(ws + 3 * F32B + 2 * B16B);
    float* attnf = qf;                              // reuse after Qb is packed

    dim3 blk(256);
    dim3 gg(HID / BN, ROWS / BM);                   // (20, 32)

    gemm_bf16<<<gg, blk, 0, stream>>>(hs, Wq, bq, qf);
    gemm_bf16<<<gg, blk, 0, stream>>>(hs, Wk, bk, kf);
    gemm_bf16<<<gg, blk, 0, stream>>>(hs, Wv, bv, vf);

    ln_rope_pack<<<ROWS, 256, 0, stream>>>(qf, qls, qlb, pos, sin_t, cos_t, Qb);
    ln_rope_pack<<<ROWS, 256, 0, stream>>>(kf, kls, klb, pos, sin_t, cos_t, Kb);

    v_pack<<<(ROWS * HID) / 256, 256, 0, stream>>>(vf, Vb);

    attention<<<dim3(SEQ / 64, 2 * NH), 128, 0, stream>>>(Qb, Kb, Vb, am, attnf);

    gemm_bf16<<<gg, blk, 0, stream>>>(attnf, Wo, bo, out);
}
